// EntInit_55035710931251
// MI455X (gfx1250) — matrix operand order repaired and checked by reading, not compiled, not run on hardware
//
#include <hip/hip_runtime.h>
#include <hip/hip_bf16.h>

// Reformulation: feat = (H @ Emb) / count, where H is the [n_nodes x 400]
// edge-type histogram (f16, exact for integer counts < 2048) and Emb is the
// combined 400x128 relation table cast to f16 (K padded to 416 = 13*32).
// Histogram built with GLOBAL_ATOMIC_PK_ADD_F16; GEMM done with
// v_wmma_f32_16x16x32_f16 (one wave per 16x16 output tile).

#define KPAD 416   // 400 rounded up to multiple of 32
#define DIM  128

typedef __attribute__((ext_vector_type(16))) _Float16 v16h;
typedef __attribute__((ext_vector_type(8)))  float    v8f;

// ---------------------------------------------------------------- zero ws
__global__ void EntInit_zero_kernel(unsigned int* hist_w, float* counts,
                                    int hist_words, int n_nodes) {
    int i = blockIdx.x * blockDim.x + threadIdx.x;
    int total = hist_words + n_nodes;
    if (i >= total) return;
    if (i < hist_words) hist_w[i] = 0u;
    else                counts[i - hist_words] = 0.0f;
}

// ------------------------------------------- build K-major f16 emb table
// embT[n][k] (n in [0,128), k in [0,KPAD)), row stride KPAD.
__global__ void EntInit_prep_kernel(const float* __restrict__ head,
                                    const float* __restrict__ tail,
                                    _Float16* __restrict__ embT) {
    int i = blockIdx.x * blockDim.x + threadIdx.x;
    if (i >= DIM * KPAD) return;
    int n = i / KPAD;
    int k = i % KPAD;
    float v = 0.0f;
    if (k < 200)      v = head[k * DIM + n];
    else if (k < 400) v = tail[(k - 200) * DIM + n];
    embT[(size_t)n * KPAD + k] = (_Float16)v;
}

// ------------------------------------------------ histogram scatter pass
__global__ void EntInit_scatter_kernel(const int* __restrict__ etypes,
                                       const int* __restrict__ dst,
                                       _Float16* __restrict__ hist,
                                       float* __restrict__ counts, int E) {
    int i = blockIdx.x * blockDim.x + threadIdx.x;
    if (i >= E) return;
    int t = etypes[i];            // 0..399  (combined table row index)
    int d = dst[i];
    // packed f16 atomic add of 1.0 into the correct half of the dword
    unsigned long long addr =
        (unsigned long long)(hist + (size_t)d * KPAD + (t & ~1));
    unsigned int val = (t & 1) ? 0x3C000000u : 0x00003C00u;  // (0,1h)/(1h,0)
    asm volatile("global_atomic_pk_add_f16 %0, %1, off scope:SCOPE_DEV"
                 :: "v"(addr), "v"(val) : "memory");
    atomicAdd(&counts[d], 1.0f);
}

// --------------------------------------------------------- WMMA GEMM
// One wave (32 lanes) per 16x16 output tile. grid.x = (n_nodes/16) * 8.
__global__ __launch_bounds__(32)
void EntInit_gemm_kernel(const _Float16* __restrict__ hist,
                         const _Float16* __restrict__ embT,
                         const float* __restrict__ counts,
                         float* __restrict__ out, int n_nodes) {
    int tile = blockIdx.x;
    int tm   = tile >> 3;          // node tile    (n_nodes/16 of them)
    int tn   = tile & 7;           // column tile  (128/16 = 8 of them)
    int lane = threadIdx.x;        // wave32
    int half = lane >> 4;          // 0: lanes 0-15, 1: lanes 16-31
    int mn   = lane & 15;

    int nodeA = tm * 16 + mn;
    if (nodeA >= n_nodes) nodeA = n_nodes - 1;       // safe clamp (exact fit anyway)
    const _Float16* arow = hist + (size_t)nodeA * KPAD;
    const _Float16* brow = embT + (size_t)(tn * 16 + mn) * KPAD;

    v8f c = {};
    #pragma unroll
    for (int kb = 0; kb < KPAD / 32; ++kb) {
        // A fragment (16x32 f16): lane holds row M=mn.
        //   elems 0..7  -> K = kb*32 + half*8 + (0..7)
        //   elems 8..15 -> K = kb*32 + half*8 + 16 + (0..7)
        int abase = kb * 32 + half * 8;
        // B fragment (32x16 f16): lane holds col N=mn.
        //   elems 0..7  -> K = kb*32 + half*8 + (0..7)
        //   elems 8..15 -> K = kb*32 + half*8 + 16 + (0..7)
        int bbase = kb * 32 + half * 8;

        union { v16h v; uint4 q[2]; } A, B;
        A.q[0] = *(const uint4*)(arow + abase);        // 8 f16, 16B aligned
        A.q[1] = *(const uint4*)(arow + abase + 16);
        B.q[0] = *(const uint4*)(brow + bbase);
        B.q[1] = *(const uint4*)(brow + bbase + 16);

        c = __builtin_amdgcn_wmma_f32_16x16x32_f16(
                /*neg_a=*/false, A.v, /*neg_b=*/false, B.v,
                /*c_mod=*/(short)0, c, /*reuse_a=*/false, /*reuse_b=*/false);
    }

    // C/D layout: VGPR v: lanes 0-15 -> M=v, lanes 16-31 -> M=v+8; N = mn.
    int ncol = tn * 16 + mn;
    #pragma unroll
    for (int v = 0; v < 8; ++v) {
        int node = tm * 16 + v + half * 8;
        if (node >= n_nodes) continue;
        float cnt = counts[node];
        float val = (cnt > 0.0f) ? (c[v] / cnt) : 0.0f;
        out[(size_t)node * DIM + ncol] = val;
    }
}

extern "C" void kernel_launch(void* const* d_in, const int* in_sizes, int n_in,
                              void* d_out, int out_size, void* d_ws, size_t ws_size,
                              hipStream_t stream) {
    const int*   etypes = (const int*)  d_in[0];
    const int*   dst    = (const int*)  d_in[1];
    const float* head   = (const float*)d_in[2];
    const float* tail   = (const float*)d_in[3];
    float*       out    = (float*)      d_out;

    const int E       = in_sizes[0];
    const int n_nodes = out_size / DIM;      // 50000

    // workspace carve-up (all 256B aligned)
    char* ws = (char*)d_ws;
    _Float16* hist = (_Float16*)ws;                        // n_nodes*KPAD f16
    size_t off = ((size_t)n_nodes * KPAD * 2 + 255) & ~(size_t)255;
    _Float16* embT = (_Float16*)(ws + off);                // 128*KPAD f16
    off += ((size_t)DIM * KPAD * 2 + 255) & ~(size_t)255;
    float* counts = (float*)(ws + off);                    // n_nodes f32

    // 1) zero hist + counts (must run every call: atomics accumulate)
    int hist_words = n_nodes * (KPAD / 2);
    {
        int total = hist_words + n_nodes;
        int blocks = (total + 255) / 256;
        EntInit_zero_kernel<<<blocks, 256, 0, stream>>>(
            (unsigned int*)hist, counts, hist_words, n_nodes);
    }
    // 2) build f16 K-major embedding table
    {
        int total = DIM * KPAD;
        int blocks = (total + 255) / 256;
        EntInit_prep_kernel<<<blocks, 256, 0, stream>>>(head, tail, embT);
    }
    // 3) histogram scatter (pk f16 atomics)
    {
        int blocks = (E + 255) / 256;
        EntInit_scatter_kernel<<<blocks, 256, 0, stream>>>(etypes, dst, hist,
                                                           counts, E);
    }
    // 4) WMMA GEMM + mean epilogue
    {
        int tilesM = (n_nodes + 15) / 16;   // 3125 (exact)
        int blocks = tilesM * (DIM / 16);   // 25000 tiles, 1 wave each
        EntInit_gemm_kernel<<<blocks, 32, 0, stream>>>(hist, embT, counts, out,
                                                       n_nodes);
    }
}
